// PolicyNetwork_17351667876579
// MI455X (gfx1250) — compile-verified, matrix operand order checked by reading, not run on hardware
//
#include <hip/hip_runtime.h>
#include <hip/hip_bf16.h>
#include <stdint.h>

// ---------------- problem constants (match reference) ----------------
#define NV   4096      // vocab
#define NS   512       // max seq len
#define NCW  8         // context window
#define NH   256       // hidden
#define ND   64        // embed dim
#define NB   65536     // batch
#define NIN  576       // CW*D + D

typedef __attribute__((ext_vector_type(16))) _Float16 v16h;
typedef __attribute__((ext_vector_type(8)))  _Float16 v8h;
typedef __attribute__((ext_vector_type(8)))  float    v8f;
typedef __attribute__((ext_vector_type(4)))  uint32_t u32x4;
typedef __attribute__((ext_vector_type(8)))  uint32_t u32x8;

#define WMMA_F16(a,b,c) \
  __builtin_amdgcn_wmma_f32_16x16x32_f16(false,(a),false,(b),(short)0,(c),false,false)

// ---------------- Tensor Data Mover: 2D f16 tile Global -> LDS ----------------
// Builds the D# per CDNA5 ISA and issues TENSOR_LOAD_TO_LDS
// via inline asm (portable across toolchains; VADDR2/3 omitted => NULL, 2D tensor).
// Tile = tensor = rows x row_elems (f16), row-major, contiguous (stride = row_elems).
// Issue from ONE wave per workgroup; EXEC is ignored by TDM. Caller must
// s_wait_tensorcnt(0) on the issuing wave, then barrier.
__device__ __forceinline__ void tdm_load_2d_f16(const void* gsrc, void* lds_dst,
                                                uint32_t rows, uint32_t row_elems) {
  union { u32x4 v; uint32_t d[4]; } g0;
  union { u32x8 v; uint32_t d[8]; } g1;
  const unsigned long long ga = (unsigned long long)(uintptr_t)gsrc;
  const uint32_t lo = (uint32_t)(uintptr_t)lds_dst;   // flat[31:0] == LDS byte offset
  // D# group 0: count=1 | lds_addr[63:32] | global_addr[120:64] | type=2 [127:126]
  g0.d[0] = 1u;
  g0.d[1] = lo;
  g0.d[2] = (uint32_t)ga;
  g0.d[3] = (uint32_t)((ga >> 32) & 0x01FFFFFFu) | (2u << 30);
  // D# group 1: mask=0 | data_size=1 (2B) | dims/strides for a 2D tile
  g1.d[0] = 1u << 16;                         // data_size = 2 bytes
  g1.d[1] = (row_elems & 0xFFFFu) << 16;      // tensor_dim0[15:0] (bits 63:48)
  g1.d[2] = (rows & 0xFFFFu) << 16;           // dim0 hi = 0 | tensor_dim1[15:0]
  g1.d[3] = (row_elems & 0xFFFFu) << 16;      // dim1 hi = 0 | tile_dim0
  g1.d[4] = rows & 0xFFFFu;                   // tile_dim1 | tile_dim2 = 0
  g1.d[5] = row_elems;                        // tensor_dim0_stride[31:0]
  g1.d[6] = 0u;                               // stride hi | tensor_dim1_stride lo
  g1.d[7] = 0u;
  asm volatile("tensor_load_to_lds %0, %1" :: "s"(g0.v), "s"(g1.v) : "memory");
}

// Load one 16x32 f16 WMMA operand tile from a row-major [rows, ld] buffer
// whose (row0, k0) element is at p0.  Per ISA 7.12.2 (16-bit A-matrix 16x32):
//   lanes 0-15 : row = lane,   K blocks {0..7, 16..23}
//   lanes16-31 : row = lane-16,K blocks {8..15,24..31}
// B operand uses the same pattern with N in place of M when W is stored
// transposed ([N,K] row-major), so one loader serves both operands.
__device__ __forceinline__ v16h ldtile(const _Float16* p0, int ld, int lane) {
  const int r  = lane & 15;
  const int ko = (lane & 16) ? 8 : 0;
  const _Float16* p = p0 + (long)r * ld + ko;
  v8h lo = *reinterpret_cast<const v8h*>(p);        // K = k0+ko   .. +7
  v8h hi = *reinterpret_cast<const v8h*>(p + 16);   // K = k0+ko+16.. +23
  return __builtin_shufflevector(lo, hi, 0,1,2,3,4,5,6,7,8,9,10,11,12,13,14,15);
}

// One wave computes a 16x32 output slab (two 16x16 WMMA tiles) of
// relu(A @ W^T + bias) and writes it f16 into LDS out[16][256].
__device__ __forceinline__ void gemm_relu_tile(
    const _Float16* A0, int lda, int K,
    const _Float16* Wt /*[256,K] row-major*/, const float* bias,
    _Float16* out /* LDS [16][256] */, int wave, int lane) {
  const int n0 = wave * 32;
  v8f acc0 = {}; v8f acc1 = {};
  for (int k = 0; k < K; k += 32) {
    v16h a  = ldtile(A0 + k, lda, lane);
    v16h b0 = ldtile(Wt + (long)n0        * K + k, K, lane);
    v16h b1 = ldtile(Wt + (long)(n0 + 16) * K + k, K, lane);
    acc0 = WMMA_F16(a, b0, acc0);
    acc1 = WMMA_F16(a, b1, acc1);
  }
  const int n  = lane & 15;
  const int mo = (lane & 16) ? 8 : 0;
  const float bb0 = bias[n0 + n], bb1 = bias[n0 + 16 + n];
#pragma unroll
  for (int r = 0; r < 8; r++) {   // acc[r] -> element (M = mo+r, N)
    out[(mo + r) * NH + n0 + n]      = (_Float16)fmaxf(acc0[r] + bb0, 0.0f);
    out[(mo + r) * NH + n0 + 16 + n] = (_Float16)fmaxf(acc1[r] + bb1, 0.0f);
  }
}

// ---------------- kernel 0a: f32 -> f16 elementwise ----------------
__global__ void k_cvt(const float* __restrict__ in, _Float16* __restrict__ out, long n) {
  long i = (long)blockIdx.x * 256 + threadIdx.x;
  if (i < n) out[i] = (_Float16)in[i];
}

// ---------------- kernel 0b: f32 [K,N] -> f16 transposed [N,K] ----------------
__global__ void k_cvt_t(const float* __restrict__ in, _Float16* __restrict__ out,
                        int K, int N) {
  long i = (long)blockIdx.x * 256 + threadIdx.x;   // i = n*K + k (coalesced store)
  if (i < (long)K * N) {
    int n = (int)(i / K), k = (int)(i % K);
    out[i] = (_Float16)in[(long)k * N + n];
  }
}

// ---------------- kernel 1: build x[B,576] f16 ----------------
__global__ void k_build_x(const int* __restrict__ seq, const int* __restrict__ pos,
                          const _Float16* __restrict__ emb16,
                          const _Float16* __restrict__ pe16,
                          _Float16* __restrict__ X) {
  const int b = blockIdx.x;           // one row per block, 64 threads
  const int t = threadIdx.x;
  const int p = pos[b];
  _Float16* xr = X + (long)b * NIN;
#pragma unroll
  for (int j = 0; j < NCW; j++) {
    int off = p - NCW + j;
    int tok = (off >= 0) ? seq[(long)b * NS + off] : 0;
    xr[j * ND + t] = emb16[(long)tok * ND + t];
  }
  xr[NCW * ND + t] = pe16[(long)p * ND + t];
}

// ---------------- kernel 2: fused layers 1+2 (policy & value) + value head ----
__global__ __launch_bounds__(256)
void k_fused12(const _Float16* __restrict__ X,
               const _Float16* __restrict__ pW1t, const float* __restrict__ pb1,
               const _Float16* __restrict__ pW2t, const float* __restrict__ pb2,
               const _Float16* __restrict__ vW1t, const float* __restrict__ vb1,
               const _Float16* __restrict__ vW2t, const float* __restrict__ vb2,
               const float* __restrict__ vW3,     const float* __restrict__ vb3,
               _Float16* __restrict__ HP, float* __restrict__ values) {
  __shared__ _Float16 xT[16 * NIN];  // 18 KB: TDM-staged A panel (shared by 8 waves)
  __shared__ _Float16 hA[16 * NH];   // ping
  __shared__ _Float16 hB[16 * NH];   // pong
  const int tid = threadIdx.x, wave = tid >> 5, lane = tid & 31;
  const long rowBase = (long)blockIdx.x * 16;

  // one TDM op DMAs the whole 16x576 x-tile into LDS (TENSORcnt-tracked)
  if (wave == 0) {
    tdm_load_2d_f16(X + rowBase * NIN, xT, 16, NIN);
    __builtin_amdgcn_s_wait_tensorcnt(0);
  }
  __syncthreads();

  // policy: h1 = relu(x W1 + b1)  (A from LDS), h2 = relu(h1 W2 + b2) (A from LDS)
  gemm_relu_tile(xT, NIN, NIN, pW1t, pb1, hA, wave, lane);
  __syncthreads();
  gemm_relu_tile(hA, NH, NH, pW2t, pb2, hB, wave, lane);
  __syncthreads();
  for (int i = tid; i < 16 * NH; i += 256)       // coalesced tile store to global
    HP[rowBase * NH + i] = hB[i];
  // value net (reuse LDS buffers; barrier below orders the HP reads of hB)
  gemm_relu_tile(xT, NIN, NIN, vW1t, vb1, hA, wave, lane);
  __syncthreads();
  gemm_relu_tile(hA, NH, NH, vW2t, vb2, hB, wave, lane);
  __syncthreads();
  if (tid < 16) {                                // value head: [256] dot per row
    float s = vb3[0];
    for (int k = 0; k < NH; k++) s += (float)hB[tid * NH + k] * vW3[k];
    values[rowBase + tid] = s;
  }
}

// ---------------- kernel 3: layer 3 (256->4096) + softmax, logits in LDS -----
__global__ __launch_bounds__(256)
void k_layer3_softmax(const _Float16* __restrict__ HP,
                      const _Float16* __restrict__ W3t /*[4096,256]*/,
                      const float* __restrict__ pb3,
                      float* __restrict__ out) {
  extern __shared__ float lds[];
  float* logits = lds;                 // [16][4096]  = 256 KB (CDNA5 320KB LDS)
  float* red    = lds + 16 * NV;       // [256] reduction scratch
  float* rowinv = red + 256;           // [16]
  _Float16* hpT = (_Float16*)(rowinv + 16);  // [16][256] TDM-staged A panel, 8 KB
  const int tid = threadIdx.x, wave = tid >> 5, lane = tid & 31;
  const long rowBase = (long)blockIdx.x * 16;

  // one TDM op DMAs the 16x256 HP panel into LDS; all waves read it from there
  if (wave == 0) {
    tdm_load_2d_f16(HP + rowBase * NH, hpT, 16, NH);
    __builtin_amdgcn_s_wait_tensorcnt(0);
  }
  __syncthreads();

  // Hold the whole 16x256 A panel in registers: 8 k-tiles, reused for 256 N-tiles.
  v16h Areg[8];
#pragma unroll
  for (int kt = 0; kt < 8; kt++) Areg[kt] = ldtile(hpT + kt * 32, NH, lane);

  const int n  = lane & 15;
  const int mo = (lane & 16) ? 8 : 0;
  for (int nt = wave; nt < NV / 16; nt += 8) {
    const int n0 = nt * 16;
    __builtin_prefetch(W3t + (long)(n0 + 128) * NH, 0, 1);   // global_prefetch_b8
    v8f acc = {};
#pragma unroll
    for (int kt = 0; kt < 8; kt++) {
      v16h b = ldtile(W3t + (long)n0 * NH + kt * 32, NH, lane);
      acc = WMMA_F16(Areg[kt], b, acc);
    }
    const float bb = pb3[n0 + n];
#pragma unroll
    for (int r = 0; r < 8; r++)
      logits[(mo + r) * NV + n0 + n] = acc[r] + bb;
  }
  __syncthreads();

  // softmax over each of 16 rows, 16 threads per row, 256 elems per thread
  const int row = tid >> 4, part = tid & 15, seg = NV / 16;
  float mx = -3.0e38f;
  for (int i = 0; i < seg; i++)
    mx = fmaxf(mx, logits[row * NV + part * seg + i]);
  red[tid] = mx;
  __syncthreads();
  float rmax = red[row * 16];
  for (int j = 1; j < 16; j++) rmax = fmaxf(rmax, red[row * 16 + j]);
  __syncthreads();                         // all rmax reads done before red reuse
  float s = 0.0f;
  for (int i = 0; i < seg; i++) {
    int idx = row * NV + part * seg + i;
    float e = __expf(logits[idx] - rmax);  // v_exp_f32 trans op
    logits[idx] = e;
    s += e;
  }
  red[tid] = s;
  __syncthreads();
  if (part == 0) {
    float rs = 0.0f;
    for (int j = 0; j < 16; j++) rs += red[row * 16 + j];
    rowinv[row] = 1.0f / rs;
  }
  __syncthreads();
  // single coalesced pass over the 1.07 GB output
  for (int i = tid; i < 16 * NV; i += 256)
    out[rowBase * NV + i] = logits[i] * rowinv[i >> 12];
}

// ---------------- host side ----------------
extern "C" void kernel_launch(void* const* d_in, const int* in_sizes, int n_in,
                              void* d_out, int out_size, void* d_ws, size_t ws_size,
                              hipStream_t stream) {
  (void)in_sizes; (void)n_in; (void)out_size; (void)ws_size;
  const int*   seq = (const int*)  d_in[0];
  const int*   pos = (const int*)  d_in[1];
  const float* emb = (const float*)d_in[2];
  const float* pe  = (const float*)d_in[3];
  const float* pW1 = (const float*)d_in[4];
  const float* pb1 = (const float*)d_in[5];
  const float* pW2 = (const float*)d_in[6];
  const float* pb2 = (const float*)d_in[7];
  const float* pW3 = (const float*)d_in[8];
  const float* pb3 = (const float*)d_in[9];
  const float* vW1 = (const float*)d_in[10];
  const float* vb1 = (const float*)d_in[11];
  const float* vW2 = (const float*)d_in[12];
  const float* vb2 = (const float*)d_in[13];
  const float* vW3 = (const float*)d_in[14];
  const float* vb3 = (const float*)d_in[15];

  float* probs  = (float*)d_out;                  // [B,V]
  float* values = probs + (size_t)NB * NV;        // [B]

  // workspace carve-up (f16 regions, 256B aligned); total ~113 MB
  char* w = (char*)d_ws;
  auto take = [&](size_t halves) -> _Float16* {
    _Float16* p = (_Float16*)w;
    w += ((halves * sizeof(_Float16) + 255) / 256) * 256;
    return p;
  };
  _Float16* emb16 = take((size_t)NV * ND);
  _Float16* pe16  = take((size_t)NS * ND);
  _Float16* pW1t  = take((size_t)NH * NIN);
  _Float16* pW2t  = take((size_t)NH * NH);
  _Float16* pW3t  = take((size_t)NV * NH);
  _Float16* vW1t  = take((size_t)NH * NIN);
  _Float16* vW2t  = take((size_t)NH * NH);
  _Float16* X     = take((size_t)NB * NIN);
  _Float16* HP    = take((size_t)NB * NH);

  auto gr = [](long n) { return (unsigned)((n + 255) / 256); };

  // 0) one-time f16 conversion / transposition of tables and weights
  k_cvt  <<<gr((long)NV * ND), 256, 0, stream>>>(emb, emb16, (long)NV * ND);
  k_cvt  <<<gr((long)NS * ND), 256, 0, stream>>>(pe,  pe16,  (long)NS * ND);
  k_cvt_t<<<gr((long)NIN * NH), 256, 0, stream>>>(pW1, pW1t, NIN, NH);
  k_cvt_t<<<gr((long)NH  * NH), 256, 0, stream>>>(pW2, pW2t, NH,  NH);
  k_cvt_t<<<gr((long)NH  * NV), 256, 0, stream>>>(pW3, pW3t, NH,  NV);
  k_cvt_t<<<gr((long)NIN * NH), 256, 0, stream>>>(vW1, vW1t, NIN, NH);
  k_cvt_t<<<gr((long)NH  * NH), 256, 0, stream>>>(vW2, vW2t, NH,  NH);

  // 1) embedding gather + PE concat
  k_build_x<<<NB, ND, 0, stream>>>(seq, pos, emb16, pe16, X);

  // 2) fused hidden layers (policy + value) and value head; TDM-staged A panel
  k_fused12<<<NB / 16, 256, 0, stream>>>(X, pW1t, pb1, pW2t, pb2,
                                         vW1t, vb1, vW2t, vb2, vW3, vb3,
                                         HP, values);

  // 3) policy head + softmax; 16x4096 f32 logits + TDM-staged HP panel in LDS
  const size_t smem = (size_t)(16 * NV + 256 + 16) * sizeof(float)
                    + (size_t)16 * NH * sizeof(_Float16);   // 271424 B
  k_layer3_softmax<<<NB / 16, 256, smem, stream>>>(HP, pW3t, pb3, probs);
}
